// BHSDuelingDQN_27187142984207
// MI455X (gfx1250) — matrix operand order repaired and checked by reading, not compiled, not run on hardware
//
#include <hip/hip_runtime.h>
#include <hip/hip_bf16.h>

typedef __attribute__((ext_vector_type(8)))  __bf16 bf16x8;
typedef __attribute__((ext_vector_type(16))) __bf16 bf16x16;
typedef __attribute__((ext_vector_type(8)))  float  floatx8;

#define BATCH    1024
#define N_NODES  256
#define F_IN     8
#define OUT_C    128
#define FEAT     32768   /* N_NODES * OUT_C */
#define NCOLS    256     /* 192 adv cols + 64 val cols */
#define ADV_COLS 192
#define NDIV     64
#define KSPLIT   4
#define KCHUNK   (FEAT / KSPLIT)          /* 8192 */
#define CSTRIDE  (BATCH * NCOLS)          /* elements per K-partial */

static __device__ __forceinline__ unsigned short f2bf(float f) {
    union { float f; unsigned u; } v; v.f = f;
    unsigned r = v.u + 0x7FFFu + ((v.u >> 16) & 1u);   // round-to-nearest-even
    return (unsigned short)(r >> 16);
}

// ---------------------------------------------------------------- prep:
// zero agg[256*128], build combined bias[256] = {adv_b, v1b}
__global__ __launch_bounds__(256) void prep_kernel(
    float* __restrict__ agg, const float* __restrict__ advb,
    const float* __restrict__ v1b, float* __restrict__ biasc)
{
    int tid = blockIdx.x * 256 + threadIdx.x;
    if (tid < N_NODES * OUT_C) agg[tid] = 0.0f;
    int b = tid - N_NODES * OUT_C;
    if (b >= 0 && b < NCOLS)
        biasc[b] = (b < ADV_COLS) ? advb[b] : v1b[b - ADV_COLS];
}

// ---------------------------------------------------------------- theta:
// edge_attr==1 for every edge -> single [F_IN,OUT_C] matrix
__global__ __launch_bounds__(128) void theta_kernel(
    const float* __restrict__ w1, const float* __restrict__ b1,
    const float* __restrict__ w2, const float* __restrict__ b2,
    float* __restrict__ theta)
{
    __shared__ float h[64];
    int t = threadIdx.x;
    if (t < 64) h[t] = fmaxf(w1[t] + b1[t], 0.0f);
    __syncthreads();
    for (int idx = t; idx < F_IN * OUT_C; idx += 128) {
        float acc = b2[idx];
        #pragma unroll 8
        for (int k = 0; k < 64; ++k) acc += h[k] * w2[k * (F_IN * OUT_C) + idx];
        theta[idx] = acc;
    }
}

// ---------------------------------------------------------------- edges:
// msg = flat[src] @ theta, scatter-add at tgt (batch-0 rows only, as in ref)
__global__ __launch_bounds__(128) void edge_kernel(
    const float* __restrict__ flat, const int* __restrict__ eidx,
    const float* __restrict__ theta, float* __restrict__ agg, int E)
{
    int e = blockIdx.x;
    if (e >= E) return;
    int o = threadIdx.x;
    int s = eidx[e];        // edge_index[0][e]
    int t = eidx[E + e];    // edge_index[1][e]
    float acc = 0.0f;
    #pragma unroll
    for (int f = 0; f < F_IN; ++f)
        acc += flat[s * F_IN + f] * theta[f * OUT_C + o];
    atomicAdd(&agg[t * OUT_C + o], acc);
}

// ---------------------------------------------------------------- conv+feat:
// feat = relu(agg(+only rows<256) + flat@root_w + conv_b), stored bf16
__global__ __launch_bounds__(256) void conv_feat_kernel(
    const float* __restrict__ flat, const float* __restrict__ rootw,
    const float* __restrict__ convb, const float* __restrict__ agg,
    unsigned int* __restrict__ featbf)  // packed 2x bf16
{
    int tid = blockIdx.x * 256 + threadIdx.x;       // one per 2 outputs
    int row = tid >> 6;                             // 64 threads/row
    if (row >= BATCH * N_NODES) return;
    int cp = (tid & 63) * 2;
    const float* xr = flat + (size_t)row * F_IN;
    float a0 = convb[cp], a1 = convb[cp + 1];
    if (row < N_NODES) { a0 += agg[row * OUT_C + cp]; a1 += agg[row * OUT_C + cp + 1]; }
    #pragma unroll
    for (int f = 0; f < F_IN; ++f) {
        float xv = xr[f];
        a0 += xv * rootw[f * OUT_C + cp];
        a1 += xv * rootw[f * OUT_C + cp + 1];
    }
    a0 = fmaxf(a0, 0.0f); a1 = fmaxf(a1, 0.0f);
    featbf[tid] = (unsigned)f2bf(a0) | ((unsigned)f2bf(a1) << 16);
}

// ---------------------------------------------------------------- wcvt:
// Wt[n][k] bf16, n<192 from adv_w[k][n], else v1w[k][n-192]  (K-contiguous)
__global__ __launch_bounds__(256) void wcvt_kernel(
    const float* __restrict__ advw, const float* __restrict__ v1w,
    unsigned int* __restrict__ Wt)   // packed 2x bf16
{
    int tid = blockIdx.x * 256 + threadIdx.x;       // one per 2 k's
    if (tid >= NCOLS * (FEAT / 2)) return;
    int n = tid >> 14;                              // FEAT/2 = 16384 pairs per n
    int k = (tid & 16383) * 2;
    float w0, w1;
    if (n < ADV_COLS) {
        w0 = advw[(size_t)k * ADV_COLS + n];
        w1 = advw[(size_t)(k + 1) * ADV_COLS + n];
    } else {
        int nn = n - ADV_COLS;
        w0 = v1w[(size_t)k * NDIV + nn];
        w1 = v1w[(size_t)(k + 1) * NDIV + nn];
    }
    Wt[tid] = (unsigned)f2bf(w0) | ((unsigned)f2bf(w1) << 16);
}

// ---------------------------------------------------------------- big GEMM:
// [1024, 32768] x [32768, 256] via v_wmma_f32_16x16x32_bf16, split-K by 4.
// Block = 8 waves (4 M x 2 N), wave tile = 16x32 (two accumulators).
// Each K-chunk writes its own partial plane of Cpart (branch-free, no atomics).
__global__ __launch_bounds__(256) void gemm_wmma_kernel(
    const __bf16* __restrict__ A,   // feat  [1024][32768]
    const __bf16* __restrict__ Bt,  // Wt    [256][32768]  (col of W, K-contig)
    float* __restrict__ Cpart)      // [KSPLIT][1024][256]
{
    const int lane = threadIdx.x & 31;
    const int wave = threadIdx.x >> 5;
    const int mi = wave & 3;         // 4 M sub-tiles
    const int ni = wave >> 2;        // 2 N sub-tiles (32 cols each)
    const int m0 = blockIdx.y * 64 + mi * 16;
    const int n0 = blockIdx.x * 64 + ni * 32;
    const int kb = blockIdx.z * KCHUNK;
    const int l15 = lane & 15;
    const int kg  = lane >> 4;

    // A fragment (contiguous k order, same as B): lane row = m0+l15,
    //   elem i -> k = k0 + 16*kg + i
    const __bf16* Ap = A + (size_t)(m0 + l15) * FEAT + kb + 16 * kg;
    // B fragment (contiguous k order, same as A): lane col = n+l15, elem i -> k = k0 + 16*kg + i
    const __bf16* B0 = Bt + (size_t)(n0 + l15) * FEAT + kb + 16 * kg;
    const __bf16* B1 = Bt + (size_t)(n0 + 16 + l15) * FEAT + kb + 16 * kg;

    floatx8 c0 = {}, c1 = {};
    #pragma unroll 4
    for (int k0 = 0; k0 < KCHUNK; k0 += 32) {
        bf16x8 alo = *(const bf16x8*)(Ap + k0);
        bf16x8 ahi = *(const bf16x8*)(Ap + k0 + 8);
        bf16x16 a = __builtin_shufflevector(alo, ahi,
            0,1,2,3,4,5,6,7,8,9,10,11,12,13,14,15);
        bf16x16 b0 = *(const bf16x16*)(B0 + k0);
        bf16x16 b1 = *(const bf16x16*)(B1 + k0);
        c0 = __builtin_amdgcn_wmma_f32_16x16x32_bf16(false, a, false, b0,
                                                     (short)0, c0, false, false);
        c1 = __builtin_amdgcn_wmma_f32_16x16x32_bf16(false, a, false, b1,
                                                     (short)0, c1, false, false);
    }

    // Branch-free epilogue: store partials.
    // C layout: col = n + l15, row = m0 + j + 8*kg
    float* Cp = Cpart + (size_t)blockIdx.z * CSTRIDE;
    int col0 = n0 + l15, col1 = n0 + 16 + l15;
    #pragma unroll
    for (int j = 0; j < 8; ++j) {
        int row = m0 + j + 8 * kg;
        Cp[(size_t)row * NCOLS + col0] = c0[j];
        Cp[(size_t)row * NCOLS + col1] = c1[j];
    }
}

// ---------------------------------------------------------------- head:
// reduce split-K partials, +bias, relu -> adv row & val1 row in LDS, then
// val2 = relu(val1@v2w+b), val3 = val2@v3w+b, out = val3 + adv - mean(adv)
__global__ __launch_bounds__(192) void head_kernel(
    const float* __restrict__ Cpart, const float* __restrict__ biasc,
    const float* __restrict__ v2w, const float* __restrict__ v2b,
    const float* __restrict__ v3w, const float* __restrict__ v3b,
    float* __restrict__ out)
{
    __shared__ float sadv[ADV_COLS], s1[64], s2[64], s3[64];
    int b = blockIdx.x, t = threadIdx.x;
    const float* C0 = Cpart + (size_t)b * NCOLS;

    // adv columns 0..191
    {
        float s = C0[t] + C0[CSTRIDE + t] + C0[2 * CSTRIDE + t] + C0[3 * CSTRIDE + t];
        sadv[t] = fmaxf(s + biasc[t], 0.0f);
    }
    // val columns 192..255
    if (t < 64) {
        int c = ADV_COLS + t;
        float s = C0[c] + C0[CSTRIDE + c] + C0[2 * CSTRIDE + c] + C0[3 * CSTRIDE + c];
        s1[t] = fmaxf(s + biasc[c], 0.0f);
    }
    __syncthreads();
    if (t < 64) {
        float acc = v2b[t];
        #pragma unroll 8
        for (int k = 0; k < 64; ++k) acc += s1[k] * v2w[k * 64 + t];
        s2[t] = fmaxf(acc, 0.0f);
    }
    __syncthreads();
    if (t < 64) {
        float acc = v3b[t];
        #pragma unroll 8
        for (int k = 0; k < 64; ++k) acc += s2[k] * v3w[k * 64 + t];
        s3[t] = acc;
    }
    __syncthreads();
    int d = t / 3;
    float a0 = sadv[d * 3], a1 = sadv[d * 3 + 1], a2 = sadv[d * 3 + 2];
    float mean = (a0 + a1 + a2) * (1.0f / 3.0f);
    out[(size_t)b * ADV_COLS + t] = s3[d] + sadv[t] - mean;
}

// ---------------------------------------------------------------- launch
extern "C" void kernel_launch(void* const* d_in, const int* in_sizes, int n_in,
                              void* d_out, int out_size, void* d_ws, size_t ws_size,
                              hipStream_t stream) {
    const float* x     = (const float*)d_in[0];
    const int*   eidx  = (const int*)  d_in[1];
    const float* w1    = (const float*)d_in[2];
    const float* b1    = (const float*)d_in[3];
    const float* w2    = (const float*)d_in[4];
    const float* b2    = (const float*)d_in[5];
    const float* rootw = (const float*)d_in[6];
    const float* convb = (const float*)d_in[7];
    const float* advw  = (const float*)d_in[8];
    const float* advb  = (const float*)d_in[9];
    const float* v1w   = (const float*)d_in[10];
    const float* v1b   = (const float*)d_in[11];
    const float* v2w   = (const float*)d_in[12];
    const float* v2b   = (const float*)d_in[13];
    const float* v3w   = (const float*)d_in[14];
    const float* v3b   = (const float*)d_in[15];
    float* out = (float*)d_out;
    int E = in_sizes[1] / 2;

    // workspace layout (bytes, 256-aligned)
    char* ws = (char*)d_ws;
    float*        theta  = (float*)(ws + 0);                  //   4 KB
    float*        agg    = (float*)(ws + 4096);               // 128 KB
    float*        biasc  = (float*)(ws + 135168);             //   1 KB
    float*        Cpart  = (float*)(ws + 136192);             //   4 MB
    unsigned int* featbf = (unsigned int*)(ws + 4330496);     //  64 MB (bf16 x2)
    unsigned int* Wt     = (unsigned int*)(ws + 71439360);    //  16 MB (bf16 x2)

    prep_kernel<<<131, 256, 0, stream>>>(agg, advb, v1b, biasc);
    theta_kernel<<<1, 128, 0, stream>>>(w1, b1, w2, b2, theta);
    edge_kernel<<<E, 128, 0, stream>>>(x, eidx, theta, agg, E);
    conv_feat_kernel<<<(BATCH * N_NODES * 64) / 256, 256, 0, stream>>>(
        x, rootw, convb, agg, featbf);
    wcvt_kernel<<<(NCOLS * FEAT / 2) / 256, 256, 0, stream>>>(advw, v1w, Wt);
    gemm_wmma_kernel<<<dim3(NCOLS / 64, BATCH / 64, KSPLIT), 256, 0, stream>>>(
        (const __bf16*)featbf, (const __bf16*)Wt, Cpart);
    head_kernel<<<BATCH, 192, 0, stream>>>(Cpart, biasc, v2w, v2b, v3w, v3b, out);
}
